// OptimizedTransformer_3693671875326
// MI455X (gfx1250) — matrix operand order repaired and checked by reading, not compiled, not run on hardware
//
#include <hip/hip_runtime.h>
#include <cstdint>
#include <cstddef>

// ---------------------------------------------------------------------------
// Performer transformer forward (B=2, N=4096, D=1024, H=16, DH=64, R=64, L=6)
// All dense GEMMs use v_wmma_f32_16x16x32_bf16 (CDNA5 wave32 WMMA).
// Weights are streamed f32 -> converted to bf16 in-register (bandwidth-bound).
// Activations kept bf16 in workspace; f32 accumulate; x lives in d_out (f32).
// ---------------------------------------------------------------------------

typedef __bf16 bf16;
typedef __attribute__((ext_vector_type(16))) __bf16 v16bf;
typedef __attribute__((ext_vector_type(8)))  float  v8f;

#define BB    2
#define NN    4096
#define DDIM  1024
#define HH    16
#define DH    64
#define RRF   64
#define LLAY  6
#define FFD   4096
#define MM    (BB*NN)     // 8192 token rows
#define BHH   (BB*HH)     // 32 (batch,head) pairs
#define LN_EPS 1e-5f
#define STAB   1e-6f

struct B32x8 { uint4 lo, hi; };   // 32 bytes == v16bf

static __device__ __forceinline__ v8f wmma_bf16(v16bf a, v16bf b, v8f c) {
  // D = A(16x32 bf16) * B(32x16 bf16) + C(16x16 f32)
  return __builtin_amdgcn_wmma_f32_16x16x32_bf16(false, a, false, b,
                                                 (short)0, c, false, false);
}

// A-fragment (16x32 bf16): lane row = lane&15; half = lane>>4.
// element e <-> K = k0 + 16*half + e (contiguous 32B)
static __device__ __forceinline__ v16bf load_a_bf16(const bf16* A, int lda,
                                                    int row, int k0, int half) {
  const bf16* p = A + (size_t)row * lda + k0 + half * 16;
  B32x8 r;
  r.lo = *(const uint4*)(p);
  r.hi = *(const uint4*)(p + 8);
  return __builtin_bit_cast(v16bf, r);
}

// B-fragment (32x16 bf16): lane col = lane&15; element e <-> K = k0+16*half+e.
// Source contiguous in K (bf16):
static __device__ __forceinline__ v16bf load_b_bf16_contig(const bf16* p) {
  B32x8 r;
  r.lo = *(const uint4*)(p);
  r.hi = *(const uint4*)(p + 8);
  return __builtin_bit_cast(v16bf, r);
}
// Source contiguous in K (f32 -> cvt bf16): p = &W[col][k0+16*half]
static __device__ __forceinline__ v16bf load_b_f32_row(const float* p) {
  v16bf b;
#pragma unroll
  for (int e = 0; e < 16; ++e) b[e] = (bf16)p[e];
  return b;
}
// Source strided in K (f32, K-major matrix [K, Ncols]): p = &B[k0+16*half][col]
static __device__ __forceinline__ v16bf load_b_f32_strided(const float* p, int stride) {
  v16bf b;
#pragma unroll
  for (int e = 0; e < 16; ++e) b[e] = (bf16)p[e * stride];
  return b;
}
static __device__ __forceinline__ v8f zero8() {
  v8f z = {0.f, 0.f, 0.f, 0.f, 0.f, 0.f, 0.f, 0.f};
  return z;
}

// ---------------------------------------------------------------------------
// Embedding: x[b,n,:] = tok_emb[tokens[b,n]] + pos_emb[n]
// ---------------------------------------------------------------------------
__global__ __launch_bounds__(256) void embed_kernel(const int* __restrict__ tokens,
                                                    const float* __restrict__ tok_emb,
                                                    const float* __restrict__ pos_emb,
                                                    float* __restrict__ x) {
  int row = blockIdx.x;            // 0..MM-1
  int n   = row & (NN - 1);
  int t   = tokens[row];
  const float* te = tok_emb + (size_t)t * DDIM;
  const float* pe = pos_emb + (size_t)n * DDIM;
  float* xr = x + (size_t)row * DDIM;
  for (int i = threadIdx.x; i < DDIM; i += 256) xr[i] = te[i] + pe[i];
}

// ---------------------------------------------------------------------------
// LayerNorm (row of 1024) -> bf16
// ---------------------------------------------------------------------------
__global__ __launch_bounds__(256) void ln_kernel(const float* __restrict__ x,
                                                 const float* __restrict__ g,
                                                 const float* __restrict__ b,
                                                 bf16* __restrict__ out) {
  __shared__ float red[16];
  int row = blockIdx.x;
  const float* xr = x + (size_t)row * DDIM;
  int wave = threadIdx.x >> 5, lane = threadIdx.x & 31;
  float s = 0.f, s2 = 0.f;
  for (int i = threadIdx.x; i < DDIM; i += 256) {
    float v = xr[i];
    s += v; s2 += v * v;
  }
#pragma unroll
  for (int o = 16; o > 0; o >>= 1) {
    s  += __shfl_xor(s, o);
    s2 += __shfl_xor(s2, o);
  }
  if (lane == 0) { red[wave] = s; red[8 + wave] = s2; }
  __syncthreads();
  if (threadIdx.x == 0) {
    float a = 0.f, c = 0.f;
#pragma unroll
    for (int w = 0; w < 8; ++w) { a += red[w]; c += red[8 + w]; }
    red[0] = a; red[8] = c;
  }
  __syncthreads();
  float mean = red[0] * (1.f / DDIM);
  float var  = red[8] * (1.f / DDIM) - mean * mean;
  float rstd = rsqrtf(var + LN_EPS);
  bf16* orow = out + (size_t)row * DDIM;
  for (int i = threadIdx.x; i < DDIM; i += 256)
    orow[i] = (bf16)((xr[i] - mean) * rstd * g[i] + b[i]);
}

// ---------------------------------------------------------------------------
// Generic linear: C[M,N] = A_bf16[M,K] @ W_f32[N,K]^T  (torch Linear layout)
// Block = 8 waves; wave -> 16(M) x 64(N) strip; block -> 128 x 64 tile.
// MODE 0: QKV scatter -> q[BH,N,DH], k[BH,N,DH], vT[BH,DH,N] (bf16)
// MODE 1: out_f32[gm,gn] += bias[gn] + acc   (residual accumulate into x)
// MODE 2: out_bf16[gm,gn] = gelu_exact(acc + bias[gn])
// ---------------------------------------------------------------------------
template <int MODE>
__global__ __launch_bounds__(256) void linear_kernel(const bf16*  __restrict__ A,
                                                     const float* __restrict__ W,
                                                     const float* __restrict__ bias,
                                                     int N, int K,
                                                     float* __restrict__ f32out,
                                                     bf16*  __restrict__ bfout,
                                                     bf16*  __restrict__ qb,
                                                     bf16*  __restrict__ kb,
                                                     bf16*  __restrict__ vT) {
  int wave = threadIdx.x >> 5;
  int lane = threadIdx.x & 31;
  int half = lane >> 4;
  int col  = lane & 15;
  int m0 = blockIdx.y * 128 + wave * 16;
  int n0 = blockIdx.x * 64;

  v8f acc[4];
#pragma unroll
  for (int t = 0; t < 4; ++t) acc[t] = zero8();

  for (int k0 = 0; k0 < K; k0 += 32) {
    v16bf a = load_a_bf16(A, K, m0 + col, k0, half);
#pragma unroll
    for (int t = 0; t < 4; ++t) {
      const float* wp = W + (size_t)(n0 + t * 16 + col) * K + k0 + half * 16;
      v16bf b = load_b_f32_row(wp);
      acc[t] = wmma_bf16(a, b, acc[t]);
    }
  }

#pragma unroll
  for (int t = 0; t < 4; ++t) {
    int gn = n0 + t * 16 + col;
#pragma unroll
    for (int r = 0; r < 8; ++r) {
      int gm = m0 + r + 8 * half;
      float v = acc[t][r];
      if (MODE == 0) {
        int b_ = gm >> 12;            // / NN
        int n_ = gm & (NN - 1);
        int which = gn >> 10;         // 0:q 1:k 2:v
        int rem = gn & 1023;
        int hh = rem >> 6;
        int d  = rem & 63;
        size_t bh = (size_t)(b_ * HH + hh);
        bf16 bv = (bf16)v;
        if (which == 0)      qb[(bh * NN + n_) * DH + d] = bv;
        else if (which == 1) kb[(bh * NN + n_) * DH + d] = bv;
        else                 vT[(bh * DH + d) * NN + n_] = bv;
      } else if (MODE == 1) {
        float* o = f32out + (size_t)gm * N + gn;
        *o = *o + bias[gn] + v;
      } else {  // MODE 2: exact GELU -> bf16
        float tt = v + bias[gn];
        float gl = 0.5f * tt * (1.0f + erff(tt * 0.70710678118654752f));
        bfout[(size_t)gm * N + gn] = (bf16)gl;
      }
    }
  }
}

// ---------------------------------------------------------------------------
// phi = relu(in[bh,N,DH] @ rf[h]) ; rf[h] is [DH,R] K-major.
// transpose==0 -> out[bh,N,R]; transpose==1 -> out[bh,R,N]
// ---------------------------------------------------------------------------
__global__ __launch_bounds__(256) void phi_kernel(const bf16* __restrict__ in,
                                                  const float* __restrict__ rf,
                                                  bf16* __restrict__ out,
                                                  int transpose) {
  int bh = blockIdx.y;
  int h  = bh & (HH - 1);
  int wave = threadIdx.x >> 5, lane = threadIdx.x & 31;
  int half = lane >> 4, col = lane & 15;
  int m0 = blockIdx.x * 128 + wave * 16;                 // n rows
  const bf16*  Ab  = in + (size_t)bh * NN * DH;
  const float* rfb = rf + (size_t)h * DH * RRF;

  v8f acc[4];
#pragma unroll
  for (int t = 0; t < 4; ++t) acc[t] = zero8();

#pragma unroll
  for (int k0 = 0; k0 < DH; k0 += 32) {
    v16bf a = load_a_bf16(Ab, DH, m0 + col, k0, half);
#pragma unroll
    for (int t = 0; t < 4; ++t) {
      const float* bp = rfb + (size_t)(k0 + 16 * half) * RRF + t * 16 + col;
      acc[t] = wmma_bf16(a, load_b_f32_strided(bp, RRF), acc[t]);
    }
  }
#pragma unroll
  for (int t = 0; t < 4; ++t) {
    int rr = t * 16 + col;
#pragma unroll
    for (int r = 0; r < 8; ++r) {
      int n_ = m0 + r + 8 * half;
      bf16 v = (bf16)fmaxf(acc[t][r], 0.f);
      if (!transpose) out[((size_t)bh * NN + n_) * RRF + rr] = v;
      else            out[((size_t)bh * RRF + rr) * NN + n_] = v;
    }
  }
}

// ksum[bh,r] = sum_n phi_kT[bh,r,n]
__global__ __launch_bounds__(64) void ksum_kernel(const bf16* __restrict__ phikT,
                                                  float* __restrict__ ksum) {
  int bh = blockIdx.x, r = threadIdx.x;
  const bf16* p = phikT + ((size_t)bh * RRF + r) * NN;
  float s = 0.f;
  for (int n = 0; n < NN; ++n) s += (float)p[n];
  ksum[bh * RRF + r] = s;
}

// kv[bh,r,d] = sum_n phi_kT[bh,r,n] * vT[bh,d,n]  (K = 4096 reduction GEMM)
// One block per (b,h); 16 waves, each owns one 16x16 tile of the 64x64 output.
__global__ __launch_bounds__(512) void kv_kernel(const bf16* __restrict__ phikT,
                                                 const bf16* __restrict__ vT,
                                                 float* __restrict__ kv) {
  int bh = blockIdx.x;
  int wave = threadIdx.x >> 5, lane = threadIdx.x & 31;
  int half = lane >> 4, col = lane & 15;
  int rt = (wave >> 2) * 16;
  int dt = (wave & 3) * 16;
  const bf16* Ab = phikT + (size_t)bh * RRF * NN;  // [R, N]
  const bf16* Bb = vT    + (size_t)bh * DH  * NN;  // [DH, N]; B[k=n][d] = vT[d][n]
  v8f acc = zero8();
  for (int k0 = 0; k0 < NN; k0 += 32) {
    v16bf a = load_a_bf16(Ab, NN, rt + col, k0, half);
    const bf16* bp = Bb + (size_t)(dt + col) * NN + k0 + 16 * half;
    acc = wmma_bf16(a, load_b_bf16_contig(bp), acc);
  }
#pragma unroll
  for (int r = 0; r < 8; ++r)
    kv[((size_t)bh * RRF + rt + r + 8 * half) * DH + dt + col] = acc[r];
}

// normz[bh,n] = phi_q[bh,n,:] . ksum[bh,:] + STAB
__global__ __launch_bounds__(256) void normz_kernel(const bf16* __restrict__ phiq,
                                                    const float* __restrict__ ksum,
                                                    float* __restrict__ normz) {
  int idx = blockIdx.x * 256 + threadIdx.x;   // bh*NN + n
  int bh = idx >> 12;                         // / NN
  const bf16*  p  = phiq + (size_t)idx * RRF;
  const float* ks = ksum + bh * RRF;
  float s = 0.f;
#pragma unroll 8
  for (int r = 0; r < RRF; ++r) s += (float)p[r] * ks[r];
  normz[idx] = s + STAB;
}

// out_heads[b,n,h*DH+d] = (phi_q[bh,n,:] @ kv[bh,:,d]) / normz[bh,n]
__global__ __launch_bounds__(256) void attn_out_kernel(const bf16* __restrict__ phiq,
                                                       const float* __restrict__ kv,
                                                       const float* __restrict__ normz,
                                                       bf16* __restrict__ out_heads) {
  int bh = blockIdx.y;
  int b_ = bh >> 4, h = bh & (HH - 1);
  int wave = threadIdx.x >> 5, lane = threadIdx.x & 31;
  int half = lane >> 4, col = lane & 15;
  int m0 = blockIdx.x * 128 + wave * 16;
  const bf16*  Ab = phiq + (size_t)bh * NN * RRF;  // [N, R]
  const float* Bb = kv   + (size_t)bh * RRF * DH;  // [R, DH] K-major

  v8f acc[4];
#pragma unroll
  for (int t = 0; t < 4; ++t) acc[t] = zero8();
#pragma unroll
  for (int k0 = 0; k0 < RRF; k0 += 32) {
    v16bf a = load_a_bf16(Ab, RRF, m0 + col, k0, half);
#pragma unroll
    for (int t = 0; t < 4; ++t) {
      const float* bp = Bb + (size_t)(k0 + 16 * half) * DH + t * 16 + col;
      acc[t] = wmma_bf16(a, load_b_f32_strided(bp, DH), acc[t]);
    }
  }
#pragma unroll
  for (int t = 0; t < 4; ++t) {
    int d = t * 16 + col;
#pragma unroll
    for (int r = 0; r < 8; ++r) {
      int n_ = m0 + r + 8 * half;
      float v = acc[t][r] / normz[bh * NN + n_];
      out_heads[((size_t)(b_ * NN + n_)) * DDIM + h * DH + d] = (bf16)v;
    }
  }
}

// ---------------------------------------------------------------------------
extern "C" void kernel_launch(void* const* d_in, const int* in_sizes, int n_in,
                              void* d_out, int out_size, void* d_ws, size_t ws_size,
                              hipStream_t stream) {
  (void)in_sizes; (void)n_in; (void)out_size; (void)ws_size;
  const int*   tokens  = (const int*)  d_in[0];
  const float* tok_emb = (const float*)d_in[1];
  const float* pos_emb = (const float*)d_in[2];
  const float* Wqkv    = (const float*)d_in[3];
  const float* Wout    = (const float*)d_in[4];
  const float* bout    = (const float*)d_in[5];
  const float* ln1_g   = (const float*)d_in[6];
  const float* ln1_b   = (const float*)d_in[7];
  const float* ln2_g   = (const float*)d_in[8];
  const float* ln2_b   = (const float*)d_in[9];
  const float* W1      = (const float*)d_in[10];
  const float* b1      = (const float*)d_in[11];
  const float* W2      = (const float*)d_in[12];
  const float* b2      = (const float*)d_in[13];
  const float* rf      = (const float*)d_in[14];

  float* x = (float*)d_out;                 // residual stream lives in d_out
  char*  ws = (char*)d_ws;

  const size_t SZ = (size_t)MM * DDIM * 2;  // 16 MiB (8192x1024 bf16)
  bf16*  hbuf  = (bf16*)(ws);
  bf16*  qb    = (bf16*)(ws + 1 * SZ);
  bf16*  kb    = (bf16*)(ws + 2 * SZ);
  bf16*  vT    = (bf16*)(ws + 3 * SZ);
  bf16*  phiq  = (bf16*)(ws + 4 * SZ);
  bf16*  phikT = (bf16*)(ws + 5 * SZ);
  bf16*  outh  = (bf16*)(ws + 6 * SZ);
  bf16*  ff1   = (bf16*)(ws + 1 * SZ);      // aliases qb..phiq (dead during FFN)
  float* kvb   = (float*)(ws + 7 * SZ);
  float* ksum  = (float*)(ws + 7 * SZ + (size_t)BHH * RRF * DH * 4);
  float* normz = (float*)(ws + 7 * SZ + (size_t)BHH * RRF * DH * 4 + (size_t)BHH * RRF * 4);

  embed_kernel<<<MM, 256, 0, stream>>>(tokens, tok_emb, pos_emb, x);

  for (int l = 0; l < LLAY; ++l) {
    const float* Wqkv_l = Wqkv + (size_t)l * 3 * DDIM * DDIM;
    const float* Wout_l = Wout + (size_t)l * DDIM * DDIM;
    const float* W1_l   = W1   + (size_t)l * FFD * DDIM;
    const float* W2_l   = W2   + (size_t)l * DDIM * FFD;
    const float* rf_l   = rf   + (size_t)l * HH * DH * RRF;

    // --- attention ---
    ln_kernel<<<MM, 256, 0, stream>>>(x, ln1_g + l * DDIM, ln1_b + l * DDIM, hbuf);
    linear_kernel<0><<<dim3(3 * DDIM / 64, MM / 128), 256, 0, stream>>>(
        hbuf, Wqkv_l, nullptr, 3 * DDIM, DDIM, nullptr, nullptr, qb, kb, vT);
    phi_kernel<<<dim3(NN / 128, BHH), 256, 0, stream>>>(qb, rf_l, phiq, 0);
    phi_kernel<<<dim3(NN / 128, BHH), 256, 0, stream>>>(kb, rf_l, phikT, 1);
    ksum_kernel<<<BHH, 64, 0, stream>>>(phikT, ksum);
    kv_kernel<<<BHH, 512, 0, stream>>>(phikT, vT, kvb);
    normz_kernel<<<BHH * NN / 256, 256, 0, stream>>>(phiq, ksum, normz);
    attn_out_kernel<<<dim3(NN / 128, BHH), 256, 0, stream>>>(phiq, kvb, normz, outh);
    linear_kernel<1><<<dim3(DDIM / 64, MM / 128), 256, 0, stream>>>(
        outh, Wout_l, bout + l * DDIM, DDIM, DDIM, x, nullptr, nullptr, nullptr, nullptr);

    // --- FFN ---
    ln_kernel<<<MM, 256, 0, stream>>>(x, ln2_g + l * DDIM, ln2_b + l * DDIM, hbuf);
    linear_kernel<2><<<dim3(FFD / 64, MM / 128), 256, 0, stream>>>(
        hbuf, W1_l, b1 + l * FFD, FFD, DDIM, nullptr, ff1, nullptr, nullptr, nullptr);
    linear_kernel<1><<<dim3(DDIM / 64, MM / 128), 256, 0, stream>>>(
        ff1, W2_l, b2 + l * DDIM, DDIM, FFD, x, nullptr, nullptr, nullptr, nullptr);
  }
}
